// MoETransformerBlock_85633057948180
// MI455X (gfx1250) — matrix operand order repaired and checked by reading, not compiled, not run on hardware
//
#include <hip/hip_runtime.h>

#define B_    2
#define S_    2048
#define D_    1024
#define H_    16
#define HD_   64
#define E_    8
#define F_    2048
#define T_    (B_ * S_)

typedef __bf16 bf16x16 __attribute__((ext_vector_type(16)));
typedef float  f32x8   __attribute__((ext_vector_type(8)));

union FragBF { bf16x16 v; unsigned int u[8]; };

__device__ __forceinline__ unsigned short f2bf(float f) {
  union { float f; unsigned int u; } x; x.f = f;
  unsigned int r = x.u + 0x7FFFu + ((x.u >> 16) & 1u);
  return (unsigned short)(r >> 16);
}
__device__ __forceinline__ float bf2f(unsigned short h) {
  union { unsigned int u; float f; } x; x.u = ((unsigned int)h) << 16;
  return x.f;
}
__device__ __forceinline__ unsigned int pack2(float lo, float hi) {
  return (unsigned int)f2bf(lo) | ((unsigned int)f2bf(hi) << 16);
}
__device__ __forceinline__ f32x8 zero8() {
  f32x8 z = {0.f, 0.f, 0.f, 0.f, 0.f, 0.f, 0.f, 0.f};
  return z;
}
__device__ __forceinline__ f32x8 wmma_bf16(const FragBF& a, const FragBF& b, f32x8 c) {
  return __builtin_amdgcn_wmma_f32_16x16x32_bf16(false, a.v, false, b.v,
                                                 (short)0, c, false, false);
}

// ---- CDNA5 async global->LDS copies (ASYNCcnt-tracked, ISA 15.18.3) --------
__device__ __forceinline__ void async_b128(void* lds, const void* g) {
  asm volatile("global_load_async_to_lds_b128 %0, %1, off"
               :: "v"((unsigned)(size_t)lds), "v"(g) : "memory");
}
__device__ __forceinline__ void async_b32(void* lds, const void* g) {
  asm volatile("global_load_async_to_lds_b32 %0, %1, off"
               :: "v"((unsigned)(size_t)lds), "v"(g) : "memory");
}
__device__ __forceinline__ void wait_async() {
  asm volatile("s_wait_asynccnt 0" ::: "memory");
}

// ---------------- Weight pack: f32 [K,N] -> K-pair-packed bf16 uint[K/2][N] -
__global__ __launch_bounds__(256) void k_pack_weights(
    const float* __restrict__ W, unsigned int* __restrict__ out, int nshift) {
  const int gid = blockIdx.x * 256 + threadIdx.x;  // over (K/2)*N
  const int N = 1 << nshift;
  const int kp = gid >> nshift, n = gid & (N - 1);
  out[gid] = pack2(W[(size_t)(2 * kp) * N + n], W[(size_t)(2 * kp + 1) * N + n]);
}

// ---------------- LayerNorm -> bf16 ----------------
__global__ __launch_bounds__(256) void k_layernorm_bf16(
    const float* __restrict__ x, const float* __restrict__ g,
    const float* __restrict__ bta, unsigned short* __restrict__ out) {
  __shared__ float s_sum[256], s_sq[256];
  const int row = blockIdx.x;
  const float* xr = x + (size_t)row * D_;
  float sum = 0.f, sq = 0.f;
  for (int i = threadIdx.x; i < D_; i += 256) { float v = xr[i]; sum += v; sq += v * v; }
  s_sum[threadIdx.x] = sum; s_sq[threadIdx.x] = sq;
  __syncthreads();
  for (int off = 128; off > 0; off >>= 1) {
    if (threadIdx.x < off) {
      s_sum[threadIdx.x] += s_sum[threadIdx.x + off];
      s_sq[threadIdx.x]  += s_sq[threadIdx.x + off];
    }
    __syncthreads();
  }
  const float mean = s_sum[0] / (float)D_;
  const float var  = s_sq[0] / (float)D_ - mean * mean;
  const float rstd = rsqrtf(var + 1e-5f);
  for (int i = threadIdx.x; i < D_; i += 256) {
    float v = (xr[i] - mean) * rstd * g[i] + bta[i];
    out[(size_t)row * D_ + i] = f2bf(v);
  }
}

// ---------------- WMMA GEMM: A bf16 [M,K] x B packed-bf16 [K/2,N] -----------
// Double-buffered async pipeline: fills for tile kt+1 overlap WMMA on tile kt.
// EPI 0: out bf16 = acc + bias
// EPI 1: out f32  = acc + bias + resid
// EPI 2: out f32 += wrow[row*E_] * (acc + bias)
template <int EPI>
__global__ __launch_bounds__(256) void k_gemm_bf16(
    const unsigned short* __restrict__ A, const unsigned int* __restrict__ Bw,
    const float* __restrict__ bias, float* __restrict__ outf,
    unsigned short* __restrict__ outh, const float* __restrict__ resid,
    const float* __restrict__ wrow, int M, int N, int Kd) {
  __shared__ unsigned short As[2][128][40];  // 80B rows (16B aligned)
  __shared__ unsigned int   Bp[2][16][68];   // 272B rows
  const int tid = threadIdx.x;
  const int lane = tid & 31, wid = tid >> 5;
  const int m0 = blockIdx.y * 128, n0 = blockIdx.x * 64;
  const int tm0 = (wid & 3) * 32, tn0 = (wid >> 2) * 32;
  f32x8 acc[2][2];
#pragma unroll
  for (int a = 0; a < 2; ++a)
#pragma unroll
    for (int b = 0; b < 2; ++b) acc[a][b] = zero8();

  auto fill = [&](int buf, int kt) {
    const int k0 = kt << 5;
#pragma unroll
    for (int it = 0; it < 2; ++it) {       // A tile: 512 x b128
      int idx = tid + it * 256;
      int r = idx >> 2, cb = idx & 3;
      async_b128(&As[buf][r][cb * 8], &A[(size_t)(m0 + r) * Kd + k0 + cb * 8]);
    }
    int kp = tid >> 4, cu = (tid & 15) * 4;  // B tile: 256 x b128
    async_b128(&Bp[buf][kp][cu], &Bw[(size_t)((k0 >> 1) + kp) * N + n0 + cu]);
  };

  const int nk = Kd >> 5;
  fill(0, 0);
  for (int kt = 0; kt < nk; ++kt) {
    const int cur = kt & 1;
    wait_async();       // drain this wave's fills for buf[cur]
    __syncthreads();    // all waves: buf[cur] resident, buf[1-cur] free
    if (kt + 1 < nk) fill(1 - cur, kt + 1);  // overlap with compute
    const int mloc = lane & 15, half = lane >> 4;
    FragBF fa[2], fb[2];
#pragma unroll
    for (int mi = 0; mi < 2; ++mi)
#pragma unroll
      for (int j = 0; j < 8; ++j) {
        int kk = half * 16 + 2 * j;
        fa[mi].u[j] = *(const unsigned int*)&As[cur][tm0 + mi * 16 + mloc][kk];
      }
    const int kb8 = half * 8;
#pragma unroll
    for (int ni = 0; ni < 2; ++ni)
#pragma unroll
      for (int j = 0; j < 8; ++j)
        fb[ni].u[j] = Bp[cur][kb8 + j][tn0 + ni * 16 + mloc];
#pragma unroll
    for (int mi = 0; mi < 2; ++mi)
#pragma unroll
      for (int ni = 0; ni < 2; ++ni)
        acc[mi][ni] = wmma_bf16(fa[mi], fb[ni], acc[mi][ni]);
  }
  const int mloc = lane & 15, half = lane >> 4;
#pragma unroll
  for (int mi = 0; mi < 2; ++mi)
#pragma unroll
    for (int ni = 0; ni < 2; ++ni)
#pragma unroll
      for (int i = 0; i < 8; ++i) {
        int row = m0 + tm0 + mi * 16 + i + half * 8;
        int col = n0 + tn0 + ni * 16 + mloc;
        float val = acc[mi][ni][i] + bias[col];
        size_t o = (size_t)row * N + col;
        if constexpr (EPI == 0) {
          outh[o] = f2bf(val);
        } else if constexpr (EPI == 1) {
          outf[o] = val + resid[o];
        } else {
          outf[o] += wrow[(size_t)row * E_] * val;
        }
      }
}

// ---------------- Fused gate/up expert GEMM, silu(g)*u epilogue -------------
__global__ __launch_bounds__(256) void k_gateup(
    const unsigned short* __restrict__ A, const unsigned int* __restrict__ Wgp,
    const unsigned int* __restrict__ Wup, const float* __restrict__ bg,
    const float* __restrict__ bu, unsigned short* __restrict__ act) {
  __shared__ unsigned short As[2][128][40];
  __shared__ unsigned int   Bpg[2][16][68];
  __shared__ unsigned int   Bpu[2][16][68];
  const int tid = threadIdx.x;
  const int lane = tid & 31, wid = tid >> 5;
  const int m0 = blockIdx.y * 128, n0 = blockIdx.x * 64;
  const int tm0 = (wid & 3) * 32, tn0 = (wid >> 2) * 32;
  f32x8 accg[2][2], accu[2][2];
#pragma unroll
  for (int a = 0; a < 2; ++a)
#pragma unroll
    for (int b = 0; b < 2; ++b) { accg[a][b] = zero8(); accu[a][b] = zero8(); }

  auto fill = [&](int buf, int kt) {
    const int k0 = kt << 5;
#pragma unroll
    for (int it = 0; it < 2; ++it) {
      int idx = tid + it * 256;
      int r = idx >> 2, cb = idx & 3;
      async_b128(&As[buf][r][cb * 8], &A[(size_t)(m0 + r) * D_ + k0 + cb * 8]);
    }
    int kp = tid >> 4, cu = (tid & 15) * 4;
    size_t go = (size_t)((k0 >> 1) + kp) * F_ + n0 + cu;
    async_b128(&Bpg[buf][kp][cu], &Wgp[go]);
    async_b128(&Bpu[buf][kp][cu], &Wup[go]);
  };

  const int nk = D_ >> 5;
  fill(0, 0);
  for (int kt = 0; kt < nk; ++kt) {
    const int cur = kt & 1;
    wait_async();
    __syncthreads();
    if (kt + 1 < nk) fill(1 - cur, kt + 1);
    const int mloc = lane & 15, half = lane >> 4;
    FragBF fa[2], fbg[2], fbu[2];
#pragma unroll
    for (int mi = 0; mi < 2; ++mi)
#pragma unroll
      for (int j = 0; j < 8; ++j) {
        int kk = half * 16 + 2 * j;
        fa[mi].u[j] = *(const unsigned int*)&As[cur][tm0 + mi * 16 + mloc][kk];
      }
    const int kb8 = half * 8;
#pragma unroll
    for (int ni = 0; ni < 2; ++ni)
#pragma unroll
      for (int j = 0; j < 8; ++j) {
        fbg[ni].u[j] = Bpg[cur][kb8 + j][tn0 + ni * 16 + mloc];
        fbu[ni].u[j] = Bpu[cur][kb8 + j][tn0 + ni * 16 + mloc];
      }
#pragma unroll
    for (int mi = 0; mi < 2; ++mi)
#pragma unroll
      for (int ni = 0; ni < 2; ++ni) {
        accg[mi][ni] = wmma_bf16(fa[mi], fbg[ni], accg[mi][ni]);
        accu[mi][ni] = wmma_bf16(fa[mi], fbu[ni], accu[mi][ni]);
      }
  }
  const int mloc = lane & 15, half = lane >> 4;
#pragma unroll
  for (int mi = 0; mi < 2; ++mi)
#pragma unroll
    for (int ni = 0; ni < 2; ++ni)
#pragma unroll
      for (int i = 0; i < 8; ++i) {
        int row = m0 + tm0 + mi * 16 + i + half * 8;
        int col = n0 + tn0 + ni * 16 + mloc;
        float gv = accg[mi][ni][i] + bg[col];
        float uv = accu[mi][ni][i] + bu[col];
        float sg = gv / (1.f + __expf(-gv));
        act[(size_t)row * F_ + col] = f2bf(sg * uv);
      }
}

// ---------------- Flash attention (causal), WMMA QK^T and PV ----------------
__global__ __launch_bounds__(256) void k_attn(
    const unsigned short* __restrict__ q, const unsigned short* __restrict__ k,
    const unsigned short* __restrict__ v, unsigned short* __restrict__ ao) {
  __shared__ unsigned short Qs[64][72];   // 144B row, 16B aligned
  __shared__ unsigned int   KTp[32][68];  // [d-pair][j] packed bf16x2
  __shared__ unsigned int   Vp[32][68];   // [j-pair][d] packed bf16x2
  __shared__ float          Sf[64][65];
  __shared__ unsigned short Ps[64][68];
  __shared__ float mrow[64], lrow[64], arow[64];
  const int tid = threadIdx.x, lane = tid & 31, wid = tid >> 5;
  const int qt = blockIdx.x, h = blockIdx.y, b = blockIdx.z;
  const int qbase = qt * 64;
  const int tmQ = (wid & 3) * 16;
  const int tnO = (wid >> 2) * 32;

#pragma unroll
  for (int it = 0; it < 2; ++it) {  // Q tile: 64 x 64 bf16 = 512 x b128, async
    int idx = tid + it * 256;
    int r = idx >> 3, cb = idx & 7;
    async_b128(&Qs[r][cb * 8],
               &q[(((size_t)(b * S_ + qbase + r)) * H_ + h) * HD_ + cb * 8]);
  }
  if (tid < 64) { mrow[tid] = -3.0e38f; lrow[tid] = 0.f; }
  f32x8 o[2] = {zero8(), zero8()};
  wait_async();
  __syncthreads();

  for (int kt = 0; kt <= qt; ++kt) {
    const int kbase = kt * 64;
#pragma unroll
    for (int it = 0; it < 8; ++it) {  // K^T tile: dword gather, async b32
      int idx = tid + it * 256;
      int d2 = idx >> 6, j = idx & 63;
      async_b32(&KTp[d2][j],
                &k[(((size_t)(b * S_ + kbase + j)) * H_ + h) * HD_ + d2 * 2]);
    }
#pragma unroll
    for (int it = 0; it < 8; ++it) {  // V tile (pack along j, strided -> VALU)
      int idx = tid + it * 256;
      int j2 = idx >> 6, d = idx & 63;
      size_t base = (((size_t)(b * S_ + kbase + j2 * 2)) * H_ + h) * HD_ + d;
      unsigned short lo = v[base];
      unsigned short hi = v[base + (size_t)H_ * HD_];
      Vp[j2][d] = (unsigned int)lo | ((unsigned int)hi << 16);
    }
    wait_async();
    __syncthreads();
    {
      const int mloc = lane & 15, half = lane >> 4;
#pragma unroll
      for (int ti = 0; ti < 2; ++ti) {  // S = (Q K^T) / 8
        const int tnS = tnO + ti * 16;
        f32x8 s = zero8();
#pragma unroll
        for (int ks = 0; ks < 2; ++ks) {
          const int k0 = ks * 32;
          FragBF fa, fb;
#pragma unroll
          for (int j = 0; j < 8; ++j) {
            int kk = half * 16 + 2 * j;
            fa.u[j] = *(const unsigned int*)&Qs[tmQ + mloc][k0 + kk];
          }
#pragma unroll
          for (int j = 0; j < 8; ++j)
            fb.u[j] = KTp[(k0 >> 1) + half * 8 + j][tnS + mloc];
          s = wmma_bf16(fa, fb, s);
        }
#pragma unroll
        for (int i = 0; i < 8; ++i)
          Sf[tmQ + i + half * 8][tnS + mloc] = s[i] * 0.125f;
      }
    }
    __syncthreads();
    if (tid < 64) {  // online softmax, one query row per thread
      const int r = tid;
      int lim = qbase + r - kbase; if (lim > 63) lim = 63;
      float mold = mrow[r];
      float rm = -3.0e38f;
      for (int c = 0; c <= lim; ++c) rm = fmaxf(rm, Sf[r][c]);
      float mnew = fmaxf(mold, rm);
      float alpha = __expf(mold - mnew);
      float ls = 0.f;
      for (int c = 0; c < 64; ++c) {
        float p = (c <= lim) ? __expf(Sf[r][c] - mnew) : 0.f;
        ls += p;
        Ps[r][c] = f2bf(p);
      }
      lrow[r] = lrow[r] * alpha + ls;
      mrow[r] = mnew;
      arow[r] = alpha;
    }
    __syncthreads();
    {
      const int mloc = lane & 15, half = lane >> 4;
#pragma unroll
      for (int ti = 0; ti < 2; ++ti)  // rescale O by alpha(row)
#pragma unroll
        for (int i = 0; i < 8; ++i)
          o[ti][i] *= arow[tmQ + i + half * 8];
#pragma unroll
      for (int ks = 0; ks < 2; ++ks) {  // O += P V
        const int k0 = ks * 32;
        FragBF fa;
#pragma unroll
        for (int j = 0; j < 8; ++j) {
          int kk = half * 16 + 2 * j;
          fa.u[j] = *(const unsigned int*)&Ps[tmQ + mloc][k0 + kk];
        }
#pragma unroll
        for (int ti = 0; ti < 2; ++ti) {
          FragBF fb;
#pragma unroll
          for (int j = 0; j < 8; ++j)
            fb.u[j] = Vp[(k0 >> 1) + half * 8 + j][tnO + ti * 16 + mloc];
          o[ti] = wmma_bf16(fa, fb, o[ti]);
        }
      }
    }
    __syncthreads();
  }
  if (tid < 64) arow[tid] = 1.0f / lrow[tid];
  __syncthreads();
  const int mloc = lane & 15, half = lane >> 4;
#pragma unroll
  for (int ti = 0; ti < 2; ++ti)
#pragma unroll
    for (int i = 0; i < 8; ++i) {
      int row = tmQ + i + half * 8;
      int col = tnO + ti * 16 + mloc;
      ao[(((size_t)(b * S_ + qbase + row)) * H_ + h) * HD_ + col] =
          f2bf(o[ti][i] * arow[row]);
    }
}

// ---------------- Router logits ----------------
__global__ __launch_bounds__(256) void k_router(
    const unsigned short* __restrict__ flat, const float* __restrict__ Wr,
    const float* __restrict__ br, float* __restrict__ logits) {
  const int gid = blockIdx.x * 256 + threadIdx.x;  // T*E threads
  const int t = gid >> 3, e = gid & 7;
  float s = br[e];
  const unsigned short* fr = flat + (size_t)t * D_;
  for (int d = 0; d < D_; ++d) s += bf2f(fr[d]) * Wr[(size_t)d * E_ + e];
  logits[gid] = s;
}

// ---------------- Softmax probs + top-2 + combine weights ----------------
__global__ __launch_bounds__(256) void k_topk(
    const float* __restrict__ logits, float* __restrict__ probs,
    float* __restrict__ ti_out, float* __restrict__ w) {
  const int t = blockIdx.x * 256 + threadIdx.x;
  if (t >= T_) return;
  float lg[E_];
  float mx = -3.0e38f;
  for (int e = 0; e < E_; ++e) { lg[e] = logits[t * E_ + e]; mx = fmaxf(mx, lg[e]); }
  float den = 0.f;
  for (int e = 0; e < E_; ++e) den += __expf(lg[e] - mx);
  for (int e = 0; e < E_; ++e) probs[(size_t)t * E_ + e] = __expf(lg[e] - mx) / den;
  int i0 = 0;
  for (int e = 1; e < E_; ++e) if (lg[e] > lg[i0]) i0 = e;
  int i1 = (i0 == 0) ? 1 : 0;
  for (int e = 0; e < E_; ++e) if (e != i0 && lg[e] > lg[i1]) i1 = e;
  float eb = __expf(lg[i1] - lg[i0]);
  float w0 = 1.f / (1.f + eb), w1 = eb / (1.f + eb);
  for (int e = 0; e < E_; ++e) w[(size_t)t * E_ + e] = 0.f;
  w[(size_t)t * E_ + i0] = w0;
  w[(size_t)t * E_ + i1] += w1;
  ti_out[(size_t)t * 2 + 0] = (float)i0;
  ti_out[(size_t)t * 2 + 1] = (float)i1;
}

extern "C" void kernel_launch(void* const* d_in, const int* in_sizes, int n_in,
                              void* d_out, int out_size, void* d_ws, size_t ws_size,
                              hipStream_t stream) {
  (void)in_sizes; (void)n_in; (void)out_size; (void)ws_size;
  const float* x    = (const float*)d_in[0];
  const float* ln1g = (const float*)d_in[1];
  const float* ln1b = (const float*)d_in[2];
  const float* Wq = (const float*)d_in[3];  const float* bq = (const float*)d_in[4];
  const float* Wk = (const float*)d_in[5];  const float* bk = (const float*)d_in[6];
  const float* Wv = (const float*)d_in[7];  const float* bv = (const float*)d_in[8];
  const float* Wo = (const float*)d_in[9];  const float* bo = (const float*)d_in[10];
  const float* ln2g = (const float*)d_in[11];
  const float* ln2b = (const float*)d_in[12];
  const float* Wr = (const float*)d_in[13]; const float* br = (const float*)d_in[14];
  const float* Wg = (const float*)d_in[15]; const float* bg = (const float*)d_in[16];
  const float* Wu = (const float*)d_in[17]; const float* bu = (const float*)d_in[18];
  const float* Wd = (const float*)d_in[19]; const float* bd = (const float*)d_in[20];

  char* ws = (char*)d_ws;
  const size_t SZ = (size_t)T_ * D_ * 2;  // one bf16 [T,D] buffer (8 MB)
  unsigned short* h    = (unsigned short*)(ws + 0 * SZ);
  unsigned short* qb   = (unsigned short*)(ws + 1 * SZ);
  unsigned short* kb   = (unsigned short*)(ws + 2 * SZ);
  unsigned short* vb   = (unsigned short*)(ws + 3 * SZ);
  unsigned short* aob  = (unsigned short*)(ws + 4 * SZ);
  unsigned short* flat = (unsigned short*)(ws + 5 * SZ);
  unsigned short* act  = (unsigned short*)(ws + 6 * SZ);   // [T,F] bf16 = 2*SZ
  float*        logits = (float*)(ws + 8 * SZ);
  float*        w      = (float*)(ws + 8 * SZ + (size_t)T_ * E_ * 4);
  // packed-weight staging (stream-ordered reuse)
  unsigned int* stW = (unsigned int*)(ws + 8 * SZ + (size_t)2 * T_ * E_ * 4);
  unsigned int* stG = stW + (size_t)(F_ / 2) * D_;   // 4 MB each
  unsigned int* stU = stG + (size_t)(D_ / 2) * F_;

  float* xo    = (float*)d_out;                 // [T,D] f32
  float* probs = xo + (size_t)T_ * D_;          // [T,E]
  float* tiout = probs + (size_t)T_ * E_;       // [T,2]

  dim3 blk(256);
  dim3 gProj(D_ / 64, T_ / 128);
  dim3 gGU(F_ / 64, T_ / 128);
  dim3 gAttn(S_ / 64, H_, B_);
  const int gPackDD = (D_ / 2) * D_ / 256;   // pack [1024,1024]
  const int gPackDF = (D_ / 2) * F_ / 256;   // pack [1024,2048]
  const int gPackFD = (F_ / 2) * D_ / 256;   // pack [2048,1024]

  k_layernorm_bf16<<<T_, blk, 0, stream>>>(x, ln1g, ln1b, h);

  k_pack_weights<<<gPackDD, blk, 0, stream>>>(Wq, stW, 10);
  k_gemm_bf16<0><<<gProj, blk, 0, stream>>>(h, stW, bq, nullptr, qb, nullptr, nullptr, T_, D_, D_);
  k_pack_weights<<<gPackDD, blk, 0, stream>>>(Wk, stW, 10);
  k_gemm_bf16<0><<<gProj, blk, 0, stream>>>(h, stW, bk, nullptr, kb, nullptr, nullptr, T_, D_, D_);
  k_pack_weights<<<gPackDD, blk, 0, stream>>>(Wv, stW, 10);
  k_gemm_bf16<0><<<gProj, blk, 0, stream>>>(h, stW, bv, nullptr, vb, nullptr, nullptr, T_, D_, D_);

  k_attn<<<gAttn, blk, 0, stream>>>(qb, kb, vb, aob);

  k_pack_weights<<<gPackDD, blk, 0, stream>>>(Wo, stW, 10);
  k_gemm_bf16<1><<<gProj, blk, 0, stream>>>(aob, stW, bo, xo, nullptr, x, nullptr, T_, D_, D_);

  k_layernorm_bf16<<<T_, blk, 0, stream>>>(xo, ln2g, ln2b, flat);
  k_router<<<(T_ * E_) / 256, blk, 0, stream>>>(flat, Wr, br, logits);
  k_topk<<<T_ / 256, blk, 0, stream>>>(logits, probs, tiout, w);

  for (int e = 0; e < E_; ++e) {
    k_pack_weights<<<gPackDF, blk, 0, stream>>>(Wg + (size_t)e * D_ * F_, stG, 11);
    k_pack_weights<<<gPackDF, blk, 0, stream>>>(Wu + (size_t)e * D_ * F_, stU, 11);
    k_gateup<<<gGU, blk, 0, stream>>>(flat, stG, stU,
                                      bg + (size_t)e * F_, bu + (size_t)e * F_, act);
    k_pack_weights<<<gPackFD, blk, 0, stream>>>(Wd + (size_t)e * F_ * D_, stW, 10);
    k_gemm_bf16<2><<<gProj, blk, 0, stream>>>(act, stW, bd + (size_t)e * D_,
                                              xo, nullptr, nullptr, w + e, T_, D_, F_);
  }
}
